// SAGPool_90623809946213
// MI455X (gfx1250) — matrix operand order repaired and checked by reading, not compiled, not run on hardware
//
#include <hip/hip_runtime.h>
#include <hip/hip_bf16.h>
#include <stdint.h>

#define BB  64
#define NN0 512
#define EE  8192
#define FIN 256
#define HH  256
#define CC  10
#define KDIM 256   // all GEMMs in this model have K == 256

typedef __attribute__((ext_vector_type(16))) __bf16 v16bf;
typedef __attribute__((ext_vector_type(8)))  float  v8f;

union FragAB { v16bf v; uint4 q[2]; };

__device__ __forceinline__ unsigned short f32_to_bf16_rne(float f) {
    union { float f; uint32_t u; } c; c.f = f;
    uint32_t u = c.u;
    return (unsigned short)((u + 0x7FFFu + ((u >> 16) & 1u)) >> 16);
}

__device__ __forceinline__ uint32_t pack2_bf16(float a, float b) {
    return (uint32_t)f32_to_bf16_rne(a) | ((uint32_t)f32_to_bf16_rne(b) << 16);
}

// ---------------- weight prep: W[K x N] f32 -> Wt[N x K] bf16 (once per layer) ----------------
__global__ void k_cvt_w_t(const float* __restrict__ W, unsigned short* __restrict__ Wt,
                          int K, int N) {
    int i = blockIdx.x * blockDim.x + threadIdx.x;
    if (i >= K * N) return;
    int k = i % K, n = i / K;
    Wt[i] = f32_to_bf16_rne(W[(size_t)k * N + n]);
}

// ---------------- WMMA bf16 GEMM: C[MxK=256 tiles] = A[Mx256]f32 @ B[256xN] ----------------
// A: f32 row-major (converted to bf16 while staging).  Bt: bf16 [N x 256] (pre-transposed).
// block = 256 threads (8 waves), tile 128x128.  Full 128x256 B panel resident in LDS.
__global__ __launch_bounds__(256) void k_gemm_bf16(
    const float* __restrict__ A, const unsigned short* __restrict__ Bt,
    float* __restrict__ C, int M, int N)
{
    __shared__ __align__(16) unsigned short lds_bt[128 * KDIM]; // 64KB, [n][k]
    __shared__ __align__(16) unsigned short lds_a[128 * 32];    // 8KB,  [m][k]

    const int t    = threadIdx.x;
    const int wave = t >> 5;
    const int lane = t & 31;
    const int l15  = lane & 15;
    const int hi   = lane >> 4;
    const int mbase = blockIdx.x * 128;
    const int nbase = blockIdx.y * 128;
    const int mrow  = wave * 16;

    // Stage entire B panel once: 128 rows(n) x 256(k) bf16, contiguous b128 copies.
    {
        int n  = t >> 1;
        int h0 = (t & 1) * 128;
        const uint4* g = (const uint4*)(Bt + (size_t)(nbase + n) * KDIM + h0);
        uint4* s = (uint4*)&lds_bt[n * KDIM + h0];
        #pragma unroll
        for (int j = 0; j < 16; j++) s[j] = g[j];
    }

    v8f acc[8];
    #pragma unroll
    for (int i = 0; i < 8; i++)
        acc[i] = v8f{0.f,0.f,0.f,0.f,0.f,0.f,0.f,0.f};

    for (int kt = 0; kt < KDIM; kt += 32) {
        // A tile: 128x32 f32 -> bf16, each thread 16 elems
        {
            int r    = t >> 1;
            int half = (t & 1) * 16;
            int grow = mbase + r;
            uint4 o0 = make_uint4(0,0,0,0), o1 = make_uint4(0,0,0,0);
            if (grow < M) {
                const float4* g = (const float4*)(A + (size_t)grow * KDIM + kt + half);
                float4 f0 = g[0], f1 = g[1], f2 = g[2], f3 = g[3];
                o0.x = pack2_bf16(f0.x, f0.y); o0.y = pack2_bf16(f0.z, f0.w);
                o0.z = pack2_bf16(f1.x, f1.y); o0.w = pack2_bf16(f1.z, f1.w);
                o1.x = pack2_bf16(f2.x, f2.y); o1.y = pack2_bf16(f2.z, f2.w);
                o1.z = pack2_bf16(f3.x, f3.y); o1.w = pack2_bf16(f3.z, f3.w);
            }
            uint4* s = (uint4*)&lds_a[r * 32 + half];
            s[0] = o0; s[1] = o1;
        }
        __syncthreads();   // covers B panel on first iteration, A tile every iteration

        // A fragment (ISA 16-bit layout): lane<16 -> K 0..7,16..23 ; lane>=16 -> K 8..15,24..31
        FragAB fa;
        fa.q[0] = *(const uint4*)&lds_a[(mrow + l15) * 32 + hi * 8];
        fa.q[1] = *(const uint4*)&lds_a[(mrow + l15) * 32 + 16 + hi * 8];

        #pragma unroll
        for (int nb = 0; nb < 8; nb++) {
            FragAB fb;  // col = nb*16 + l15, K 0..7,16..23 (lane<16) / 8..15,24..31 (lane>=16)
            const uint4* pb = (const uint4*)&lds_bt[(nb * 16 + l15) * KDIM + kt + hi * 8];
            fb.q[0] = pb[0]; fb.q[1] = pb[2];
            acc[nb] = __builtin_amdgcn_wmma_f32_16x16x32_bf16(
                false, fa.v, false, fb.v, (short)0, acc[nb], false, false);
        }
        __syncthreads();   // protect lds_a before next staging
    }

    // C layout: VGPR r -> M = r (lanes 0-15) / 8+r (lanes 16-31), N = lane&15
    #pragma unroll
    for (int nb = 0; nb < 8; nb++) {
        int col = nbase + nb * 16 + l15;
        #pragma unroll
        for (int r = 0; r < 8; r++) {
            int row = mbase + mrow + hi * 8 + r;
            if (row < M) C[(size_t)row * N + col] = acc[nb][r];
        }
    }
}

// ---------------- graph kernels ----------------
__global__ void k_init_emask(int* emask) {
    int i = blockIdx.x * blockDim.x + threadIdx.x;
    if (i < BB * EE) emask[i] = 1;
}

__global__ void k_deg(const int* __restrict__ edges, const int* __restrict__ emask,
                      float* __restrict__ deg) {
    int i = blockIdx.x * blockDim.x + threadIdx.x;
    if (i >= BB * EE) return;
    if (!emask[i]) return;
    int dst = edges[2 * i + 1];
    int b = i / EE;
    atomicAdd(&deg[b * NN0 + dst], 1.0f);
}

__global__ void k_dinv(const float* __restrict__ deg, float* __restrict__ dinv, int n) {
    int i = blockIdx.x * blockDim.x + threadIdx.x;
    if (i < n) dinv[i] = rsqrtf(1.0f + deg[i]);
}

__global__ void k_agg(const int* __restrict__ edges, const int* __restrict__ emask,
                      const float* __restrict__ dinv, const float* __restrict__ h,
                      float* __restrict__ agg, int N) {
    const int CH = HH / 4;                 // 64 float4 chunks
    long i = (long)blockIdx.x * blockDim.x + threadIdx.x;
    if (i >= (long)BB * EE * CH) return;
    int ch = (int)(i % CH);
    long be = i / CH;
    int e = (int)(be % EE);
    int b = (int)(be / EE);
    int ei = b * EE + e;
    if (!emask[ei]) return;
    int src = edges[2 * ei], dst = edges[2 * ei + 1];
    float norm = dinv[b * NN0 + src] * dinv[b * NN0 + dst];
    const float4 hv = *(const float4*)&h[((size_t)(b * N + src)) * HH + ch * 4];
    float* o = &agg[((size_t)(b * N + dst)) * HH + ch * 4];
    atomicAdd(o + 0, hv.x * norm);
    atomicAdd(o + 1, hv.y * norm);
    atomicAdd(o + 2, hv.z * norm);
    atomicAdd(o + 3, hv.w * norm);
}

__global__ void k_finalize(const float* __restrict__ hW, const float* __restrict__ dinv,
                           const float* __restrict__ bias, float* __restrict__ agg, int N) {
    size_t i = (size_t)blockIdx.x * blockDim.x + threadIdx.x;
    size_t tot = (size_t)BB * N * HH;
    if (i >= tot) return;
    int f = (int)(i % HH);
    size_t bn = i / HH;
    int n = (int)(bn % N);
    int b = (int)(bn / N);
    float di = dinv[b * NN0 + n];
    float v = agg[i] + hW[i] * di * di + bias[f];
    agg[i] = v > 0.f ? v : 0.f;
}

__global__ void k_score_node(const float* __restrict__ h, const float* __restrict__ Wrel,
                             const float* __restrict__ Wroot, const float* __restrict__ pb,
                             float* __restrict__ tsrc, float* __restrict__ score, int N) {
    int i = blockIdx.x * blockDim.x + threadIdx.x;
    if (i >= BB * N) return;
    int n = i % N, b = i / N;
    const float* hp = &h[(size_t)(b * N + n) * HH];
    float a = 0.f, c = 0.f;
    for (int f = 0; f < HH; f++) { float x = hp[f]; a += x * Wrel[f]; c += x * Wroot[f]; }
    tsrc[b * NN0 + n]  = a;
    score[b * NN0 + n] = c + pb[0];
}

__global__ void k_score_edge(const int* __restrict__ edges, const int* __restrict__ emask,
                             const float* __restrict__ tsrc, float* __restrict__ score) {
    int i = blockIdx.x * blockDim.x + threadIdx.x;
    if (i >= BB * EE) return;
    if (!emask[i]) return;
    int b = i / EE;
    int src = edges[2 * i], dst = edges[2 * i + 1];
    atomicAdd(&score[b * NN0 + dst], tsrc[b * NN0 + src]);
}

// per-batch bitonic top-k (N is a power of two <= 512)
__global__ __launch_bounds__(512) void k_pool_sort(const float* __restrict__ score,
                                                   int N, int k, int* __restrict__ perm,
                                                   int* __restrict__ nodemap) {
    __shared__ float s[512];
    __shared__ int   id[512];
    int b = blockIdx.x, t = threadIdx.x;
    if (t < N) { s[t] = score[b * NN0 + t]; id[t] = t; }
    __syncthreads();
    for (int size = 2; size <= N; size <<= 1) {
        for (int stride = size >> 1; stride > 0; stride >>= 1) {
            if (t < N / 2) {
                int pos = 2 * t - (t & (stride - 1));
                int p2 = pos + stride;
                bool desc = ((pos & size) == 0);
                float a = s[pos], c = s[p2];
                bool sw = desc ? (a < c) : (a > c);
                if (sw) {
                    s[pos] = c; s[p2] = a;
                    int tmp = id[pos]; id[pos] = id[p2]; id[p2] = tmp;
                }
            }
            __syncthreads();
        }
    }
    if (t < N) nodemap[b * NN0 + t] = -1;
    __syncthreads();
    if (t < k) { perm[b * NN0 + t] = id[t]; nodemap[b * NN0 + id[t]] = t; }
}

__global__ void k_pool_gather(const float* __restrict__ h, const float* __restrict__ score,
                              const int* __restrict__ perm, float* __restrict__ xn,
                              int N, int k) {
    size_t i = (size_t)blockIdx.x * blockDim.x + threadIdx.x;
    size_t tot = (size_t)BB * k * HH;
    if (i >= tot) return;
    int f = (int)(i % HH);
    size_t bi = i / HH;
    int r = (int)(bi % k);
    int b = (int)(bi / k);
    int src = perm[b * NN0 + r];
    float sc = tanhf(score[b * NN0 + src]);
    xn[i] = h[(size_t)(b * N + src) * HH + f] * sc;
}

__global__ void k_edge_remap(int* __restrict__ edges, int* __restrict__ emask,
                             const int* __restrict__ nodemap) {
    int i = blockIdx.x * blockDim.x + threadIdx.x;
    if (i >= BB * EE) return;
    int b = i / EE;
    int src = edges[2 * i], dst = edges[2 * i + 1];
    int ns = nodemap[b * NN0 + src], nd = nodemap[b * NN0 + dst];
    int valid = (emask[i] != 0) && (ns >= 0) && (nd >= 0);
    edges[2 * i]     = ns > 0 ? ns : 0;
    edges[2 * i + 1] = nd > 0 ? nd : 0;
    emask[i] = valid;
}

// ---------------- head ----------------
__global__ void k_mean(const float* __restrict__ x, float* __restrict__ g, int N) {
    int i = blockIdx.x * blockDim.x + threadIdx.x;
    if (i >= BB * HH) return;
    int f = i % HH, b = i / HH;
    float s = 0.f;
    for (int n = 0; n < N; n++) s += x[(size_t)(b * N + n) * HH + f];
    g[i] = s / (float)N;
}

__global__ void k_bias_relu(float* __restrict__ x, const float* __restrict__ bias, int rows) {
    int i = blockIdx.x * blockDim.x + threadIdx.x;
    if (i >= rows * HH) return;
    float v = x[i] + bias[i % HH];
    x[i] = v > 0.f ? v : 0.f;
}

__global__ void k_head(const float* __restrict__ g, const float* __restrict__ W2,
                       const float* __restrict__ b2, float* __restrict__ out) {
    int b = blockIdx.x * blockDim.x + threadIdx.x;
    if (b >= BB) return;
    float logit[CC];
    float mx = -1e30f;
    for (int c = 0; c < CC; c++) {
        float s = b2[c];
        for (int f = 0; f < HH; f++) s += g[b * HH + f] * W2[f * CC + c];
        logit[c] = s;
        mx = fmaxf(mx, s);
    }
    float se = 0.f;
    for (int c = 0; c < CC; c++) se += expf(logit[c] - mx);
    float lse = mx + logf(se);
    for (int c = 0; c < CC; c++) out[b * CC + c] = logit[c] - lse;
}

// ---------------- host ----------------
static inline int cdiv(long a, int b) { return (int)((a + b - 1) / b); }

extern "C" void kernel_launch(void* const* d_in, const int* in_sizes, int n_in,
                              void* d_out, int out_size, void* d_ws, size_t ws_size,
                              hipStream_t stream) {
    const float* x       = (const float*)d_in[0];
    const int*   edges_i = (const int*)d_in[1];
    const float* convW[3] = { (const float*)d_in[2], (const float*)d_in[4], (const float*)d_in[6] };
    const float* convB[3] = { (const float*)d_in[3], (const float*)d_in[5], (const float*)d_in[7] };
    const float* pWrel[3] = { (const float*)d_in[8],  (const float*)d_in[11], (const float*)d_in[14] };
    const float* pWroot[3]= { (const float*)d_in[9],  (const float*)d_in[12], (const float*)d_in[15] };
    const float* pB[3]    = { (const float*)d_in[10], (const float*)d_in[13], (const float*)d_in[16] };
    const float* lin1W = (const float*)d_in[17];
    const float* lin1B = (const float*)d_in[18];
    const float* lin2W = (const float*)d_in[19];
    const float* lin2B = (const float*)d_in[20];
    float* out = (float*)d_out;

    // workspace carve (256B aligned)
    char* w = (char*)d_ws;
    auto carve = [&](size_t bytes) { char* p = w; w += (bytes + 255) & ~(size_t)255; return p; };
    unsigned short* wbf = (unsigned short*)carve((size_t)HH * KDIM * 2); // transposed bf16 weights
    float* bufA   = (float*)carve((size_t)BB * NN0 * HH * 4);  // GEMM out -> pooled x
    float* bufB   = (float*)carve((size_t)BB * NN0 * HH * 4);  // agg -> post-relu h
    float* deg    = (float*)carve((size_t)BB * NN0 * 4);
    float* dinv   = (float*)carve((size_t)BB * NN0 * 4);
    float* score  = (float*)carve((size_t)BB * NN0 * 4);
    float* tsrc   = (float*)carve((size_t)BB * NN0 * 4);
    int*   perm   = (int*)carve((size_t)BB * NN0 * 4);
    int*   nodemap= (int*)carve((size_t)BB * NN0 * 4);
    int*   edgesc = (int*)carve((size_t)BB * EE * 2 * 4);
    int*   emask  = (int*)carve((size_t)BB * EE * 4);
    float* g      = (float*)carve((size_t)BB * HH * 4);

    const int T = 256;
    hipMemcpyAsync(edgesc, edges_i, (size_t)BB * EE * 2 * sizeof(int),
                   hipMemcpyDeviceToDevice, stream);
    k_init_emask<<<cdiv((long)BB * EE, T), T, 0, stream>>>(emask);

    const float* xcur = x;
    int Ns[3] = { NN0, NN0 / 2, NN0 / 4 };   // 512, 256, 128
    for (int l = 0; l < 3; l++) {
        int N = Ns[l];
        int k = N / 2;
        long M = (long)BB * N;

        // h = xcur @ W  (bf16 WMMA GEMM, f32 accumulate; weights pre-transposed to [N][K])
        k_cvt_w_t<<<cdiv((long)KDIM * HH, T), T, 0, stream>>>(convW[l], wbf, KDIM, HH);
        dim3 gg(cdiv(M, 128), HH / 128);
        k_gemm_bf16<<<gg, T, 0, stream>>>(xcur, wbf, bufA, (int)M, HH);

        // GCN aggregation
        hipMemsetAsync(bufB, 0, (size_t)M * HH * 4, stream);
        hipMemsetAsync(deg, 0, (size_t)BB * NN0 * 4, stream);
        k_deg<<<cdiv((long)BB * EE, T), T, 0, stream>>>(edgesc, emask, deg);
        k_dinv<<<cdiv((long)BB * NN0, T), T, 0, stream>>>(deg, dinv, BB * NN0);
        k_agg<<<cdiv((long)BB * EE * (HH / 4), T), T, 0, stream>>>(edgesc, emask, dinv, bufA, bufB, N);
        k_finalize<<<cdiv(M * HH, T), T, 0, stream>>>(bufA, dinv, convB[l], bufB, N);

        // SAG score + top-k pool
        k_score_node<<<cdiv(M, T), T, 0, stream>>>(bufB, pWrel[l], pWroot[l], pB[l], tsrc, score, N);
        k_score_edge<<<cdiv((long)BB * EE, T), T, 0, stream>>>(edgesc, emask, tsrc, score);
        k_pool_sort<<<BB, 512, 0, stream>>>(score, N, k, perm, nodemap);
        k_pool_gather<<<cdiv((long)BB * k * HH, T), T, 0, stream>>>(bufB, score, perm, bufA, N, k);
        k_edge_remap<<<cdiv((long)BB * EE, T), T, 0, stream>>>(edgesc, emask, nodemap);

        xcur = bufA;   // pooled features, [BB*k, HH]
    }

    // readout: mean over 64 nodes, lin1 (WMMA) + relu, lin2 + log_softmax
    k_mean<<<cdiv((long)BB * HH, T), T, 0, stream>>>(bufA, g, NN0 / 8);  // 64 nodes
    k_cvt_w_t<<<cdiv((long)KDIM * HH, T), T, 0, stream>>>(lin1W, wbf, KDIM, HH);
    dim3 gh(cdiv(BB, 128), HH / 128);
    k_gemm_bf16<<<gh, T, 0, stream>>>(g, wbf, bufB, BB, HH);
    k_bias_relu<<<cdiv((long)BB * HH, T), T, 0, stream>>>(bufB, lin1B, BB);
    k_head<<<cdiv((long)BB, 64), 64, 0, stream>>>(bufB, lin2W, lin2B, out);
}
